// MambaClassifier_70171175682301
// MI455X (gfx1250) — matrix operand order repaired and checked by reading, not compiled, not run on hardware
//
#include <hip/hip_runtime.h>
#include <hip/hip_bf16.h>

typedef __attribute__((ext_vector_type(16))) __bf16 v16bf;
typedef __attribute__((ext_vector_type(8)))  __bf16 v8bf;
typedef __attribute__((ext_vector_type(8)))  float  v8f;

#define BB 64
#define TT 1024
#define FF 512
#define HH 512
#define MM (BB * TT)   // 65536 rows of flattened [B*T, F]

union AFrag { v16bf v; v8bf h[2]; };

// A-operand fragment: 16x32 bf16 tile, row-major source with leading dim `ld`.
// Lane layout (ISA 7.12.2): lanes 0-15 hold K=[0..7] and K=[16..23] of row M=lane;
// lanes 16-31 hold K=[8..15] and K=[24..31] of row M=lane-16.
__device__ __forceinline__ v16bf frag_A_rm(const __bf16* __restrict__ base,
                                           int ld, int k0, int lane) {
  const int hf = lane >> 4, r = lane & 15;
  const __bf16* p = base + (size_t)r * ld + k0 + hf * 8;
  AFrag f;
  f.h[0] = *(const v8bf*)(p);
  f.h[1] = *(const v8bf*)(p + 16);
  return f.v;
}

// B-operand fragment: 32x16 bf16 (KxN). Source W stored [N rows, K cols] row-major,
// so element (k,n) = W[n*ld + k]. Lane holds column n=lane&15, K chunks by lane-half:
// lanes 0-15 -> K=[k0..k0+7],[k0+16..k0+23]; lanes 16-31 -> K=[k0+8..k0+15],[k0+24..k0+31].
__device__ __forceinline__ v16bf frag_B_nk(const __bf16* __restrict__ W,
                                           int ld, int n0, int k0, int lane) {
  const int hf = lane >> 4, n = lane & 15;
  const __bf16* p = W + (size_t)(n0 + n) * ld + k0 + hf * 8;
  AFrag f;
  f.h[0] = *(const v8bf*)(p);
  f.h[1] = *(const v8bf*)(p + 16);
  return f.v;
}

__device__ __forceinline__ v8f wmma_bf16(v16bf a, v16bf b, v8f c) {
  return __builtin_amdgcn_wmma_f32_16x16x32_bf16(false, a, false, b, (short)0, c,
                                                 false, false);
}

__device__ __forceinline__ float sigmoidf_(float x) {
  return 1.0f / (1.0f + __expf(-x));
}

// XOR swizzle on 16-byte chunks so phase-B ds_load_b128 across 16 rows hits
// 16 distinct bank groups (row stride 1024B would otherwise alias mod 256B).
#define HOFF(row, col) \
  ((row) * 512 + (((((col) >> 3) ^ ((row) & 15))) << 3) + ((col) & 7))

// ---------------------------------------------------------------------------
__global__ void cvt_f32_bf16(const float* __restrict__ s, __bf16* __restrict__ d,
                             int n) {
  int i = blockIdx.x * blockDim.x + threadIdx.x;
  int st = gridDim.x * blockDim.x;
  for (; i < n; i += st) d[i] = (__bf16)s[i];
}

// ---------------------------------------------------------------------------
// Fused: h = sigmoid(x Wsel^T + bsel) * tanh(x Wenc^T + benc) (kept in LDS),
// then uh = h Wu^T streamed to global as bf16. One WG = 64 rows x full H.
__global__ __launch_bounds__(256, 1)
void encode_u_kernel(const __bf16* __restrict__ xbf,
                     const __bf16* __restrict__ wsel,
                     const __bf16* __restrict__ wenc,
                     const __bf16* __restrict__ wu,
                     const float* __restrict__ bsel,
                     const float* __restrict__ benc,
                     __bf16* __restrict__ uh) {
  __shared__ __align__(16) __bf16 hsm[64 * 512];   // 64KB swizzled h tile
  const int w    = threadIdx.x >> 5;
  const int lane = threadIdx.x & 31;
  const int hf   = lane >> 4;
  const int nl   = lane & 15;
  const size_t m0 = (size_t)blockIdx.x * 64;

  // ---- phase A: gated encoding into LDS --------------------------------
  for (int nsub = 0; nsub < 4; ++nsub) {
    const int n0 = w * 64 + nsub * 16;
    v8f aS[4], aE[4];
#pragma unroll
    for (int i = 0; i < 4; ++i) {
      aS[i] = (v8f){0, 0, 0, 0, 0, 0, 0, 0};
      aE[i] = (v8f){0, 0, 0, 0, 0, 0, 0, 0};
    }
    for (int kb = 0; kb < FF; kb += 32) {
      v16bf bS = frag_B_nk(wsel, FF, n0, kb, lane);
      v16bf bE = frag_B_nk(wenc, FF, n0, kb, lane);
#pragma unroll
      for (int i = 0; i < 4; ++i) {
        v16bf a = frag_A_rm(xbf + (m0 + 16 * i) * FF, FF, kb, lane);
        aS[i] = wmma_bf16(a, bS, aS[i]);
        aE[i] = wmma_bf16(a, bE, aE[i]);
      }
    }
    const int n = n0 + nl;
    const float bs = bsel[n], be = benc[n];
#pragma unroll
    for (int i = 0; i < 4; ++i) {
#pragma unroll
      for (int r = 0; r < 8; ++r) {
        int row = 16 * i + r + 8 * hf;
        float g = sigmoidf_(aS[i][r] + bs);
        float e = tanhf(aE[i][r] + be);
        hsm[HOFF(row, n)] = (__bf16)(g * e);
      }
    }
  }
  __syncthreads();

  // ---- phase B: uh = h Wu^T, A-fragments straight from LDS --------------
  for (int nsub = 0; nsub < 4; ++nsub) {
    const int n0 = w * 64 + nsub * 16;
    v8f acc[4];
#pragma unroll
    for (int i = 0; i < 4; ++i) acc[i] = (v8f){0, 0, 0, 0, 0, 0, 0, 0};
    for (int kb = 0; kb < HH; kb += 32) {
      v16bf b = frag_B_nk(wu, HH, n0, kb, lane);
#pragma unroll
      for (int i = 0; i < 4; ++i) {
        int row = 16 * i + nl;
        AFrag a;
        a.h[0] = *(const v8bf*)&hsm[HOFF(row, kb + 8 * hf)];
        a.h[1] = *(const v8bf*)&hsm[HOFF(row, kb + 16 + 8 * hf)];
        acc[i] = wmma_bf16(a.v, b, acc[i]);
      }
    }
    const int n = n0 + nl;
#pragma unroll
    for (int i = 0; i < 4; ++i) {
#pragma unroll
      for (int r = 0; r < 8; ++r) {
        int row = 16 * i + r + 8 * hf;
        uh[(m0 + row) * HH + n] = (__bf16)acc[i][r];
      }
    }
  }
}

// ---------------------------------------------------------------------------
// Sequential scan: k_{t+1} = sigmoid(k_t Wk^T + uh[:,t,:] + state_b).
// 16 persistent WGs, grid barrier per step, bf16 ping-pong k in L2.
__global__ __launch_bounds__(256, 1)
void scan_kernel(const __bf16* __restrict__ uh,
                 const __bf16* __restrict__ wk,
                 const float* __restrict__ stateb,
                 __bf16* __restrict__ kbuf0,
                 __bf16* __restrict__ kbuf1,
                 unsigned* __restrict__ bar) {
  const int w    = threadIdx.x >> 5;
  const int lane = threadIdx.x & 31;
  const int hf   = lane >> 4;
  const int nl   = lane & 15;
  const int m0   = (w >> 1) * 16;                 // 4 row tiles cover B=64
  const int n0   = blockIdx.x * 32 + (w & 1) * 16;
  const int n    = n0 + nl;
  const float sb = stateb[n];
  const int nwg  = gridDim.x;
  __bf16* kb[2] = {kbuf0, kbuf1};

  for (int t = 0; t < TT; ++t) {
    const __bf16* kold = kb[t & 1];
    __bf16* knew       = kb[(t + 1) & 1];
    v8f acc = (v8f){0, 0, 0, 0, 0, 0, 0, 0};
    for (int kk = 0; kk < HH; kk += 32) {
      v16bf a = frag_A_rm(kold + (size_t)m0 * HH, HH, kk, lane);
      v16bf b = frag_B_nk(wk, HH, n0, kk, lane);
      acc = wmma_bf16(a, b, acc);
    }
#pragma unroll
    for (int r = 0; r < 8; ++r) {
      int m = m0 + r + 8 * hf;                              // batch index
      float u = (float)uh[((size_t)m * TT + t) * HH + n];
      float v = sigmoidf_(acc[r] + u + sb);
      knew[(size_t)m * HH + n] = (__bf16)v;
      if (t + 1 < TT)
        __builtin_prefetch(&uh[((size_t)m * TT + (t + 1)) * HH + n], 0, 1);
    }
    // ---- device-wide step barrier (release writes, acquire reads) -------
    __threadfence();
    __syncthreads();
    if (threadIdx.x == 0) {
      unsigned prev = __hip_atomic_fetch_add(&bar[0], 1u, __ATOMIC_ACQ_REL,
                                             __HIP_MEMORY_SCOPE_AGENT);
      if (prev == (unsigned)(nwg * (t + 1) - 1)) {
        __hip_atomic_store(&bar[1], (unsigned)(t + 1), __ATOMIC_RELEASE,
                           __HIP_MEMORY_SCOPE_AGENT);
      } else {
        while (__hip_atomic_load(&bar[1], __ATOMIC_ACQUIRE,
                                 __HIP_MEMORY_SCOPE_AGENT) < (unsigned)(t + 1)) {
          __builtin_amdgcn_s_sleep(2);
        }
      }
    }
    __syncthreads();
    __threadfence();
  }
}

// ---------------------------------------------------------------------------
__global__ void cls_kernel(const __bf16* __restrict__ kfin,
                           const float* __restrict__ wcls,
                           const float* __restrict__ bcls,
                           float* __restrict__ out) {
  int tid = threadIdx.x;
  if (tid >= BB * 2) return;
  int b = tid >> 1, c = tid & 1;
  float s = 0.f;
  for (int h = 0; h < HH; ++h)
    s += (float)kfin[b * HH + h] * wcls[c * HH + h];
  out[b * 2 + c] = s + bcls[c];
}

// ---------------------------------------------------------------------------
extern "C" void kernel_launch(void* const* d_in, const int* in_sizes, int n_in,
                              void* d_out, int out_size, void* d_ws, size_t ws_size,
                              hipStream_t stream) {
  (void)in_sizes; (void)n_in; (void)out_size; (void)ws_size;
  const float* x    = (const float*)d_in[0];
  const float* Wsel = (const float*)d_in[1];
  const float* bsel = (const float*)d_in[2];
  const float* Wenc = (const float*)d_in[3];
  const float* benc = (const float*)d_in[4];
  const float* Wk   = (const float*)d_in[5];
  const float* Wu   = (const float*)d_in[6];
  const float* sb   = (const float*)d_in[7];
  const float* Wcls = (const float*)d_in[8];
  const float* bcls = (const float*)d_in[9];
  float* out = (float*)d_out;

  char* ws = (char*)d_ws;
  size_t off = 0;
  auto alloc = [&](size_t bytes) -> void* {
    void* p = ws + off;
    off = (off + bytes + 255) & ~(size_t)255;
    return p;
  };
  __bf16* xbf   = (__bf16*)alloc((size_t)MM * FF * 2);   // 64 MB
  __bf16* wselb = (__bf16*)alloc((size_t)HH * FF * 2);
  __bf16* wencb = (__bf16*)alloc((size_t)HH * FF * 2);
  __bf16* wub   = (__bf16*)alloc((size_t)HH * HH * 2);
  __bf16* wkb   = (__bf16*)alloc((size_t)HH * HH * 2);
  __bf16* uh    = (__bf16*)alloc((size_t)MM * HH * 2);   // 64 MB
  __bf16* k0b   = (__bf16*)alloc((size_t)BB * HH * 2);
  __bf16* k1b   = (__bf16*)alloc((size_t)BB * HH * 2);
  unsigned* bar = (unsigned*)alloc(256);

  hipMemsetAsync(k0b, 0, (size_t)BB * HH * 2, stream);   // k_0 = 0
  hipMemsetAsync(bar, 0, 256, stream);                   // barrier counters

  cvt_f32_bf16<<<4096, 256, 0, stream>>>(x, xbf, MM * FF);
  cvt_f32_bf16<<<256, 256, 0, stream>>>(Wsel, wselb, HH * FF);
  cvt_f32_bf16<<<256, 256, 0, stream>>>(Wenc, wencb, HH * FF);
  cvt_f32_bf16<<<256, 256, 0, stream>>>(Wu, wub, HH * HH);
  cvt_f32_bf16<<<256, 256, 0, stream>>>(Wk, wkb, HH * HH);

  encode_u_kernel<<<MM / 64, 256, 0, stream>>>(xbf, wselb, wencb, wub, bsel,
                                               benc, uh);
  scan_kernel<<<16, 256, 0, stream>>>(uh, wkb, sb, k0b, k1b, bar);
  // T=1024 steps: final k lands in kbuf[(1023+1)&1] == k0b
  cls_kernel<<<1, 128, 0, stream>>>(k0b, Wcls, bcls, out);
}
